// GPSPNAHybrid_84207128805733
// MI455X (gfx1250) — matrix operand order repaired and checked by reading, not compiled, not run on hardware
//
#include <hip/hip_runtime.h>
#include <math.h>

// ---------------- problem constants (fixed by the reference) ----------------
#define NBG   128          // graphs
#define NPER  30           // nodes per graph
#define NN    3840         // total nodes
#define CC    256          // channels
#define TT    4            // PNA towers
#define FID   64           // per-tower channels
#define EE    111360       // edges
#define NHEADS 8
#define HD2   128          // half channels
#define AVG_LIN 14.5f
#define AVG_LOG 2.488607878294339f   // mean(log(deg+1)) over uniform hist 0..29
#define APAD  40           // LDS row stride in halves (80B, 16B-aligned, conflict-free)

typedef __attribute__((ext_vector_type(16))) _Float16 v16h;
typedef __attribute__((ext_vector_type(8)))  _Float16 v8h;
typedef __attribute__((ext_vector_type(4)))  _Float16 h4;
typedef __attribute__((ext_vector_type(8)))  float    v8f;
typedef __attribute__((ext_vector_type(4)))  float    f4;

__device__ __forceinline__ float gelu_f(float x) {
  return 0.5f * x * (1.0f + erff(x * 0.70710678118654752f));
}
__device__ __forceinline__ int imin(int a, int b) { return a < b ? a : b; }

__device__ __forceinline__ void atomicMaxF(float* a, float v) {
  if (v >= 0.0f) atomicMax((int*)a, __float_as_int(v));
  else           atomicMin((unsigned int*)a, __float_as_uint(v));
}
__device__ __forceinline__ void atomicMinF(float* a, float v) {
  if (v >= 0.0f) atomicMin((int*)a, __float_as_int(v));
  else           atomicMax((unsigned int*)a, __float_as_uint(v));
}

// ---------------- generic WMMA GEMM with fused gathers ----------------
// GATHER 0: plain row-major A
// GATHER 1: PNA pre1 : A-row e -> [xh[dst[e]][toff..+64], xh[src[e]][toff..+64], e_enc[e][0..64]]
// GATHER 2: EdgeConv : A-row e -> [xh[dst[e]][0..256], (xh[src[e]]-xh[dst[e]])[0..256]]
template<int GATHER>
__device__ __forceinline__ float loadA1(const float* __restrict__ A, int row, int col,
                                        int lda, int toff,
                                        const int* __restrict__ gd,
                                        const int* __restrict__ gs,
                                        const float* __restrict__ gx) {
  if (GATHER == 0) return A[(size_t)row * lda + col];
  if (GATHER == 1) {
    if (col < FID)     return A[(size_t)gd[row] * lda + toff + col];
    if (col < 2 * FID) return A[(size_t)gs[row] * lda + toff + (col - FID)];
    return gx[(size_t)row * FID + (col - 2 * FID)];
  }
  if (col < CC) return A[(size_t)gd[row] * lda + col];
  int c = col - CC;
  return A[(size_t)gs[row] * lda + c] - A[(size_t)gd[row] * lda + c];
}

// 4-wide gather; col is a multiple of 4 and never straddles a segment boundary
template<int GATHER>
__device__ __forceinline__ f4 loadA4(const float* __restrict__ A, int row, int col,
                                     int lda, int toff,
                                     const int* __restrict__ gd,
                                     const int* __restrict__ gs,
                                     const float* __restrict__ gx) {
  if (GATHER == 0) return *(const f4*)&A[(size_t)row * lda + col];
  if (GATHER == 1) {
    if (col < FID)     return *(const f4*)&A[(size_t)gd[row] * lda + toff + col];
    if (col < 2 * FID) return *(const f4*)&A[(size_t)gs[row] * lda + toff + (col - FID)];
    return *(const f4*)&gx[(size_t)row * FID + (col - 2 * FID)];
  }
  if (col < CC) return *(const f4*)&A[(size_t)gd[row] * lda + col];
  int c = col - CC;
  f4 s = *(const f4*)&A[(size_t)gs[row] * lda + c];
  f4 d = *(const f4*)&A[(size_t)gd[row] * lda + c];
  return s - d;
}

// EPI: 0 none, 1 relu, 2 gelu, 3 BN(scale/shift)+gelu   (all applied after +bias)
// Block = 128 threads = 4 waves; block tile 128(M) x 32(N); per-wave tile 32x32
// => 4 WMMAs per k-step per wave (2 A-frags x 2 B-frags), 2 ds_load_b128 per WMMA.
template<int GATHER, int EPI>
__global__ __launch_bounds__(128)
void gemm_wmma_k(const float* __restrict__ Abase, const float* __restrict__ Bbase,
                 const float* __restrict__ bias, float* __restrict__ Cbase,
                 int M, int N, int K, int lda, int ldb, int ldc,
                 int sAz, int sBz, int sCz, int sbz,
                 const int* __restrict__ gd, const int* __restrict__ gs,
                 const float* __restrict__ gx,
                 const float* __restrict__ esc, const float* __restrict__ esh) {
  const int z    = blockIdx.z;
  const float* A = Abase + ((GATHER == 0) ? (size_t)z * sAz : (size_t)0);
  const int toff = (GATHER == 1) ? z * sAz : 0;
  const float* B = Bbase + (size_t)z * sBz;
  const float* bi = bias + (size_t)z * sbz;
  float* C = Cbase + (size_t)z * sCz;

  const int tid  = threadIdx.x;
  const int wave = tid >> 5;
  const int lane = tid & 31;
  const int row0 = (blockIdx.x * 4 + wave) * 32;   // 32 rows per wave
  const int n0   = blockIdx.y * 32;

  __shared__ _Float16 As[4][32][APAD];   // per-wave A tile 32x32 (cols 0..31 used)
  __shared__ _Float16 BsT[32][APAD];     // B transposed [n][k]

  const int am   = lane & 15;            // A row within 16-row frag
  const int koff = (lane >> 4) * 8;      // ISA 16-bit A-layout k offset
  const int bk0  = (lane >> 4) * 8;      // ISA 16-bit B-layout k offset
  const int bn   = lane & 15;            // B col within 16-col group

  v8f acc00 = {}, acc01 = {}, acc10 = {}, acc11 = {};

  for (int k0 = 0; k0 < K; k0 += 32) {
    const bool fullK = (k0 + 32 <= K);
    const bool fullA = fullK && (row0 + 32 <= M);   // wave-uniform
    const bool fullB = fullK && (n0 + 32 <= N);     // block-uniform

    // ---- stage A tile 32x32 (per wave: 256 f4-groups, 8 per lane) ----
    if (fullA) {
#pragma unroll
      for (int j = 0; j < 8; ++j) {
        int g = lane + 32 * j;
        int r = g >> 3, c4 = (g & 7) * 4;
        f4 v = loadA4<GATHER>(A, row0 + r, k0 + c4, lda, toff, gd, gs, gx);
        h4 hv;
#pragma unroll
        for (int i = 0; i < 4; ++i) hv[i] = (_Float16)v[i];
        *(h4*)&As[wave][r][c4] = hv;                // ds_store_b64
      }
    } else {
#pragma unroll
      for (int j = 0; j < 8; ++j) {
        int g = lane + 32 * j;
        int r = g >> 3, c4 = (g & 7) * 4;
        int rr = row0 + r;
        int rcl = imin(rr, M - 1);
        float tv[4];
#pragma unroll
        for (int i = 0; i < 4; ++i)
          tv[i] = loadA1<GATHER>(A, rcl, imin(k0 + c4 + i, K - 1), lda, toff, gd, gs, gx);
        h4 hv;
#pragma unroll
        for (int i = 0; i < 4; ++i)
          hv[i] = (_Float16)((rr < M && (k0 + c4 + i) < K) ? tv[i] : 0.0f);
        *(h4*)&As[wave][r][c4] = hv;
      }
    }

    // ---- stage B tile 32x32 transposed (cooperative: 256 f4-groups, 2 per thread) ----
    if (fullB) {
#pragma unroll
      for (int j = 0; j < 2; ++j) {
        int g = tid + 128 * j;
        int kk = g >> 3, n4 = (g & 7) * 4;
        f4 v = *(const f4*)&B[(size_t)(k0 + kk) * ldb + (n0 + n4)];
#pragma unroll
        for (int i = 0; i < 4; ++i) BsT[n4 + i][kk] = (_Float16)v[i];
      }
    } else {
#pragma unroll
      for (int j = 0; j < 2; ++j) {
        int g = tid + 128 * j;
        int kk = g >> 3, n4 = (g & 7) * 4;
        float tv[4];
#pragma unroll
        for (int i = 0; i < 4; ++i)
          tv[i] = B[(size_t)imin(k0 + kk, K - 1) * ldb + imin(n0 + n4 + i, N - 1)];
#pragma unroll
        for (int i = 0; i < 4; ++i)
          BsT[n4 + i][kk] = (_Float16)(((k0 + kk) < K && (n0 + n4 + i) < N) ? tv[i] : 0.0f);
      }
    }
    if (k0 + 32 < K) __builtin_prefetch(&B[(size_t)(k0 + 32) * ldb + n0], 0, 1);
    __syncthreads();

    // ---- fragments via wide LDS reads ----
    v8h a0l = *(const v8h*)&As[wave][am][koff];       // ds_load_b128
    v8h a0h = *(const v8h*)&As[wave][am][16 + koff];
    v16h av0 = __builtin_shufflevector(a0l, a0h, 0, 1, 2, 3, 4, 5, 6, 7,
                                       8, 9, 10, 11, 12, 13, 14, 15);
    v8h a1l = *(const v8h*)&As[wave][16 + am][koff];
    v8h a1h = *(const v8h*)&As[wave][16 + am][16 + koff];
    v16h av1 = __builtin_shufflevector(a1l, a1h, 0, 1, 2, 3, 4, 5, 6, 7,
                                       8, 9, 10, 11, 12, 13, 14, 15);
    v8h b0l = *(const v8h*)&BsT[bn][bk0];
    v8h b0h = *(const v8h*)&BsT[bn][bk0 + 16];
    v16h bv0 = __builtin_shufflevector(b0l, b0h, 0, 1, 2, 3, 4, 5, 6, 7,
                                       8, 9, 10, 11, 12, 13, 14, 15);
    v8h b1l = *(const v8h*)&BsT[bn + 16][bk0];
    v8h b1h = *(const v8h*)&BsT[bn + 16][bk0 + 16];
    v16h bv1 = __builtin_shufflevector(b1l, b1h, 0, 1, 2, 3, 4, 5, 6, 7,
                                       8, 9, 10, 11, 12, 13, 14, 15);

    acc00 = __builtin_amdgcn_wmma_f32_16x16x32_f16(false, av0, false, bv0, (short)0, acc00,
                                                   false, false);
    acc01 = __builtin_amdgcn_wmma_f32_16x16x32_f16(false, av0, false, bv1, (short)0, acc01,
                                                   false, false);
    acc10 = __builtin_amdgcn_wmma_f32_16x16x32_f16(false, av1, false, bv0, (short)0, acc10,
                                                   false, false);
    acc11 = __builtin_amdgcn_wmma_f32_16x16x32_f16(false, av1, false, bv1, (short)0, acc11,
                                                   false, false);
    __syncthreads();
  }

  // ---- epilogue: bias + activation, 2x2 sub-tiles ----
#pragma unroll
  for (int t2 = 0; t2 < 2; ++t2) {
    int nc = n0 + t2 * 16 + bn;
    if (nc < N) {
      float bv = bi[nc];
      float sc = 1.0f, sh = 0.0f;
      if (EPI == 3) { sc = esc[nc] * rsqrtf(1.0f + 1e-5f); sh = esh[nc]; }
#pragma unroll
      for (int r2 = 0; r2 < 2; ++r2) {
#pragma unroll
        for (int i = 0; i < 8; ++i) {
          int gr = row0 + r2 * 16 + i + ((lane >> 4) << 3);
          if (gr < M) {
            float v;
            if (r2 == 0) v = (t2 ? acc01[i] : acc00[i]);
            else         v = (t2 ? acc11[i] : acc10[i]);
            v += bv;
            if (EPI == 1)      v = fmaxf(v, 0.0f);
            else if (EPI == 2) v = gelu_f(v);
            else if (EPI == 3) v = gelu_f(v * sc + sh);
            C[(size_t)gr * ldc + nc] = v;
          }
        }
      }
    }
  }
}

// ---------------- elementwise / reduction kernels ----------------
__global__ void k_fill(float* p, float v, int n) {
  int i = blockIdx.x * 256 + threadIdx.x;
  if (i < n) p[i] = v;
}
__global__ void k_copy(const float* __restrict__ s, float* __restrict__ d, int n) {
  int i = blockIdx.x * 256 + threadIdx.x;
  if (i < n) d[i] = s[i];
}

// LayerNorm over last dim D with flags: 1 = pre-add residual, 2 = GELU, 4 = post-add residual
__global__ void k_layernorm(const float* __restrict__ in, const float* __restrict__ g,
                            const float* __restrict__ b, float* __restrict__ out,
                            const float* __restrict__ pre, const float* __restrict__ post,
                            int D, int flags) {
  __shared__ float sbuf[256];
  int row = blockIdx.x, t = threadIdx.x;
  float v = in[(size_t)row * D + t];
  if (flags & 1) v += pre[(size_t)row * D + t];
  sbuf[t] = v; __syncthreads();
  for (int s = D >> 1; s > 0; s >>= 1) { if (t < s) sbuf[t] += sbuf[t + s]; __syncthreads(); }
  float mean = sbuf[0] / (float)D; __syncthreads();
  float dv = v - mean;
  sbuf[t] = dv * dv; __syncthreads();
  for (int s = D >> 1; s > 0; s >>= 1) { if (t < s) sbuf[t] += sbuf[t + s]; __syncthreads(); }
  float var = sbuf[0] / (float)D;
  float y = dv * rsqrtf(var + 1e-5f) * g[t] + b[t];
  if (flags & 2) y = gelu_f(y);
  if (flags & 4) y += post[(size_t)row * D + t];
  out[(size_t)row * D + t] = y;
}

// edge_attr[E,3] @ W[3,Nf] + b  (K=3: VALU is right, WMMA would waste 90% of K)
__global__ void k_edge_encode(const float* __restrict__ ea, const float* __restrict__ W,
                              const float* __restrict__ b, float* __restrict__ out, int Nf) {
  int i = blockIdx.x * 256 + threadIdx.x;
  if (i >= EE * Nf) return;
  int e = i / Nf, c = i % Nf;
  out[i] = ea[e * 3] * W[c] + ea[e * 3 + 1] * W[Nf + c] + ea[e * 3 + 2] * W[2 * Nf + c] + b[c];
}

__global__ void k_deg(const int* __restrict__ dst, float* __restrict__ deg) {
  int e = blockIdx.x * 256 + threadIdx.x;
  if (e < EE) atomicAdd(&deg[dst[e]], 1.0f);
}

__global__ void k_pna_scatter(const float* __restrict__ m, const int* __restrict__ dst,
                              float* ssum, float* ssq, float* smx, float* smn) {
  int i = blockIdx.x * 256 + threadIdx.x;
  if (i >= EE * CC) return;
  int e = i >> 8, c = i & 255;
  float v = m[i];
  int o = dst[e] * CC + c;
  atomicAdd(&ssum[o], v);
  atomicAdd(&ssq[o], v * v);
  atomicMaxF(&smx[o], v);
  atomicMinF(&smn[o], v);
}

// build o_in [N, T, 1600] = [xt | agg | agg*dl/AVGLOG | agg*AVGLOG/dl | agg*deg/AVGLIN]
__global__ void k_pna_build(const float* __restrict__ xh, const float* __restrict__ ssum,
                            const float* __restrict__ ssq, const float* __restrict__ smx,
                            const float* __restrict__ smn, const float* __restrict__ deg,
                            float* __restrict__ o_out) {
  int i = blockIdx.x * 256 + threadIdx.x;
  if (i >= NN * TT * 1600) return;
  int n = i / 6400;
  int r = i % 6400;
  int t = r / 1600;
  int c = r % 1600;
  float val;
  if (c < FID) {
    val = xh[n * CC + t * FID + c];
  } else {
    int c2 = c - FID;
    int grp = c2 / 384;
    int w   = c2 % 384;
    int st  = w / FID;
    int f   = w % FID;
    int o   = n * CC + t * FID + f;
    float dg = fmaxf(deg[n], 1.0f);
    float s = ssum[o];
    float mean = s / dg;
    float var = ssq[o] / dg - mean * mean;
    float v;
    if (st == 0)      v = mean;
    else if (st == 1) v = smx[o];
    else if (st == 2) v = smn[o];
    else if (st == 3) v = sqrtf(fmaxf(var, 0.0f) + 1e-5f);
    else if (st == 4) v = var;
    else              v = s;
    float dl = logf(dg + 1.0f);
    float sc = (grp == 0) ? 1.0f
             : (grp == 1) ? (dl / AVG_LOG)
             : (grp == 2) ? (AVG_LOG / dl)
                          : (dg / AVG_LIN);
    val = v * sc;
  }
  o_out[i] = val;
}

// GINE message: x_in[dst] += relu(xh[src] + edge_attr@geW + geb)   (K=3 fused)
__global__ void k_gine(const float* __restrict__ xh, const float* __restrict__ ea,
                       const float* __restrict__ geW, const float* __restrict__ geb,
                       const int* __restrict__ src, const int* __restrict__ dst,
                       float* __restrict__ xin) {
  int i = blockIdx.x * 256 + threadIdx.x;
  if (i >= EE * CC) return;
  int e = i >> 8, c = i & 255;
  float v = xh[(size_t)src[e] * CC + c]
          + ea[e * 3] * geW[c] + ea[e * 3 + 1] * geW[CC + c] + ea[e * 3 + 2] * geW[2 * CC + c]
          + geb[c];
  v = fmaxf(v, 0.0f);
  atomicAdd(&xin[(size_t)dst[e] * CC + c], v);
}

__global__ void k_concat(const float* __restrict__ a, const float* __restrict__ b,
                         float* __restrict__ out) {
  int i = blockIdx.x * 256 + threadIdx.x;
  if (i >= NN * 512) return;
  int n = i / 512, c = i % 512;
  out[i] = (c < CC) ? a[n * CC + c] : b[n * CC + (c - CC)];
}

// per-graph MHA (S=30, 8 heads, Dh=32): one thread per query, one block per (graph, head)
__global__ void k_graph_attn(const float* __restrict__ q, const float* __restrict__ k,
                             const float* __restrict__ v, float* __restrict__ out) {
  int g = blockIdx.x, h = blockIdx.y;
  int t = threadIdx.x;
  if (t >= NPER) return;
  const float scale = rsqrtf(32.0f);
  int base = g * NPER * CC + h * 32;
  float s[NPER];
  float mx = -1e30f;
  for (int j = 0; j < NPER; ++j) {
    float a = 0.0f;
    for (int d = 0; d < 32; ++d) a += q[base + t * CC + d] * k[base + j * CC + d];
    s[j] = a * scale;
    mx = fmaxf(mx, s[j]);
  }
  float se = 0.0f;
  for (int j = 0; j < NPER; ++j) { s[j] = __expf(s[j] - mx); se += s[j]; }
  float inv = 1.0f / se;
  for (int d = 0; d < 32; ++d) {
    float a = 0.0f;
    for (int j = 0; j < NPER; ++j) a += s[j] * v[base + j * CC + d];
    out[base + t * CC + d] = a * inv;
  }
}

// temporal MHA (S=3840, 4 heads, Dh=32): block per (query, head), online softmax over keys
__global__ __launch_bounds__(128)
void k_temporal_attn(const float* __restrict__ q, const float* __restrict__ k,
                     const float* __restrict__ v, float* __restrict__ out) {
  int qi = blockIdx.x, h = blockIdx.y;
  int tid = threadIdx.x;
  const float scale = rsqrtf(32.0f);
  float qv[32];
  for (int d = 0; d < 32; ++d) qv[d] = q[(size_t)qi * HD2 + h * 32 + d];
  float m = -1e30f, ssum = 0.0f, acc[32];
  for (int d = 0; d < 32; ++d) acc[d] = 0.0f;
  for (int j = tid; j < NN; j += 128) {
    const f4* k4 = (const f4*)&k[(size_t)j * HD2 + h * 32];
    float s = 0.0f;
#pragma unroll
    for (int w = 0; w < 8; ++w) {
      f4 kv = k4[w];
#pragma unroll
      for (int i = 0; i < 4; ++i) s += qv[w * 4 + i] * kv[i];
    }
    s *= scale;
    float nm = fmaxf(m, s);
    float f = __expf(m - nm), e = __expf(s - nm);
    ssum = ssum * f + e;
    const f4* v4 = (const f4*)&v[(size_t)j * HD2 + h * 32];
#pragma unroll
    for (int w = 0; w < 8; ++w) {
      f4 vv = v4[w];
#pragma unroll
      for (int i = 0; i < 4; ++i) acc[w * 4 + i] = acc[w * 4 + i] * f + e * vv[i];
    }
    m = nm;
  }
  __shared__ float sm[128], ss[128], sa[128][32];
  sm[tid] = m; ss[tid] = ssum;
  for (int d = 0; d < 32; ++d) sa[tid][d] = acc[d];
  __syncthreads();
  for (int st = 64; st > 0; st >>= 1) {
    if (tid < st) {
      float m1 = sm[tid], m2 = sm[tid + st];
      float nm = fmaxf(m1, m2);
      float f1 = __expf(m1 - nm), f2 = __expf(m2 - nm);
      sm[tid] = nm;
      ss[tid] = ss[tid] * f1 + ss[tid + st] * f2;
      for (int d = 0; d < 32; ++d) sa[tid][d] = sa[tid][d] * f1 + sa[tid + st][d] * f2;
    }
    __syncthreads();
  }
  if (tid < 32) out[(size_t)qi * HD2 + h * 32 + tid] = sa[0][tid] / ss[0];
}

__global__ void k_edgemax(const float* __restrict__ he, const int* __restrict__ dst,
                          float* __restrict__ xd) {
  int i = blockIdx.x * 256 + threadIdx.x;
  if (i >= EE * HD2) return;
  int e = i >> 7, c = i & 127;
  atomicMaxF(&xd[(size_t)dst[e] * HD2 + c], he[i]);
}

__global__ void k_final(const float* __restrict__ f2, const float* __restrict__ mW,
                        const float* __restrict__ mb, const float* __restrict__ temp,
                        float* __restrict__ out) {
  int n = blockIdx.x * 256 + threadIdx.x;
  if (n >= NN) return;
  float a = 0.0f;
  for (int d = 0; d < 32; ++d) a += f2[n * 32 + d] * mW[d];
  out[n] = (a + mb[0]) / temp[0];
}

// ---------------- host-side GEMM launcher ----------------
template<int G, int E_>
static inline void GEMM(hipStream_t s, const float* A, const float* B, const float* bi,
                        float* C, int M, int N, int K, int lda, int ldb, int ldc,
                        int sAz, int sBz, int sCz, int sbz, int Z,
                        const int* gd = nullptr, const int* gs = nullptr,
                        const float* gx = nullptr,
                        const float* esc = nullptr, const float* esh = nullptr) {
  dim3 grid((M + 127) / 128, (N + 31) / 32, Z);
  gemm_wmma_k<G, E_><<<grid, 128, 0, s>>>(A, B, bi, C, M, N, K, lda, ldb, ldc,
                                          sAz, sBz, sCz, sbz, gd, gs, gx, esc, esh);
}

// parameter order: flattened setup_inputs() dict in insertion order
enum {
  IN_X = 0, IN_EI, IN_EA, IN_BATCH,
  P_IN_W, P_IN_B, P_IN_G, P_IN_BT,
  P_PE_W, P_PE_B,
  P_PRE1_W, P_PRE1_B, P_PRE2_W, P_PRE2_B,
  P_POST1_W, P_POST1_B, P_POST2_W, P_POST2_B,
  P_PNA_LIN_W, P_PNA_LIN_B, P_PNAN_G, P_PNAN_B,
  P_GE_W, P_GE_B, P_GN1_W, P_GN1_B, P_GN2_W, P_GN2_B,
  P_AWQ, P_ABQ, P_AWK, P_ABK, P_AWV, P_ABV, P_AWO, P_ABO,
  P_CMB1_W, P_CMB1_B, P_CMB2_W, P_CMB2_B,
  P_GPSN_G, P_GPSN_B, P_GPSN2_G, P_GPSN2_B,
  P_EC1_W, P_EC1_B, P_ECBN_G, P_ECBN_B, P_EC2_W, P_EC2_B,
  P_DN_G, P_DN_B,
  P_TWQ, P_TBQ, P_TWK, P_TBK, P_TWV, P_TBV, P_TWO, P_TBO,
  P_VH1_W, P_VH1_B, P_VH2_W, P_VH2_B, P_MEAN_W, P_MEAN_B, P_TEMP
};

extern "C" void kernel_launch(void* const* d_in, const int* in_sizes, int n_in,
                              void* d_out, int out_size, void* d_ws, size_t ws_size,
                              hipStream_t stream) {
  (void)in_sizes; (void)n_in; (void)out_size; (void)ws_size;
  auto P = [&](int i) { return (const float*)d_in[i]; };
  const float* x  = P(IN_X);
  const int*   ei = (const int*)d_in[IN_EI];
  const float* ea = P(IN_EA);
  const int* src = ei;         // edge_index[0]
  const int* dst = ei + EE;    // edge_index[1]
  float* out = (float*)d_out;

  // ---------- workspace layout (floats), with phase reuse ----------
  float* ws = (float*)d_ws;
  size_t off = 0;
  auto F = [&](size_t n) { float* p = ws + off; off += n; return p; };
  float* xh   = F((size_t)NN * CC);
  float* nb1  = F((size_t)NN * CC);
  float* nb2  = F((size_t)NN * CC);
  float* nb3  = F((size_t)NN * CC);
  float* nb4  = F((size_t)NN * CC);
  float* nb5  = F((size_t)NN * CC);
  float* nb6  = F((size_t)NN * CC);
  float* t512a = F((size_t)NN * 512);
  float* t512b = F((size_t)NN * 512);
  float* ssum = F((size_t)NN * CC);
  float* ssq  = F((size_t)NN * CC);
  float* smx  = F((size_t)NN * CC);
  float* smn  = F((size_t)NN * CC);
  float* deg  = F((size_t)NN);
  float* mA   = F((size_t)EE * CC);   // edge buf #1 (PNA pre1 out; EdgeConv hidden)
  float* mB   = F((size_t)EE * CC);   // edge buf #2 (e_enc -> m -> o_in -> he, phased)
  float* e_enc = mB;                  // [E,64]   (dead after pre1)
  float* o_in  = mB;                  // [N,T,1600] (after aggregation consumed m)
  float* he    = mB;                  // [E,128]  (after o_in consumed)

  auto GR = [](long n) { return dim3((unsigned)((n + 255) / 256)); };

  // ---------- input projection ----------
  GEMM<0, 0>(stream, x, P(P_IN_W), P(P_IN_B), nb1, NN, CC, 84, 84, CC, CC, 0, 0, 0, 0, 1);
  k_layernorm<<<NN, CC, 0, stream>>>(nb1, P(P_IN_G), P(P_IN_BT), xh, nullptr, nullptr, CC, 2);

  // ---------- PNA ----------
  k_edge_encode<<<GR((long)EE * FID), 256, 0, stream>>>(ea, P(P_PE_W), P(P_PE_B), e_enc, FID);
  GEMM<1, 1>(stream, xh, P(P_PRE1_W), P(P_PRE1_B), mA, EE, FID, 3 * FID,
             CC, FID, CC, FID, 3 * FID * FID, FID, FID, TT, dst, src, e_enc);
  GEMM<0, 0>(stream, mA, P(P_PRE2_W), P(P_PRE2_B), mB, EE, FID, FID,
             CC, FID, CC, FID, FID * FID, FID, FID, TT);
  k_fill<<<GR((long)NN * CC), 256, 0, stream>>>(ssum, 0.0f, NN * CC);
  k_fill<<<GR((long)NN * CC), 256, 0, stream>>>(ssq, 0.0f, NN * CC);
  k_fill<<<GR((long)NN * CC), 256, 0, stream>>>(smx, -3.0e38f, NN * CC);
  k_fill<<<GR((long)NN * CC), 256, 0, stream>>>(smn, 3.0e38f, NN * CC);
  k_fill<<<GR(NN), 256, 0, stream>>>(deg, 0.0f, NN);
  k_deg<<<GR(EE), 256, 0, stream>>>(dst, deg);
  k_pna_scatter<<<GR((long)EE * CC), 256, 0, stream>>>(mB, dst, ssum, ssq, smx, smn);
  k_pna_build<<<GR((long)NN * TT * 1600), 256, 0, stream>>>(xh, ssum, ssq, smx, smn, deg, o_in);
  GEMM<0, 1>(stream, o_in, P(P_POST1_W), P(P_POST1_B), nb1, NN, FID, 25 * FID,
             TT * 1600, FID, CC, 1600, 25 * FID * FID, FID, FID, TT);
  GEMM<0, 0>(stream, nb1, P(P_POST2_W), P(P_POST2_B), nb2, NN, FID, FID,
             CC, FID, CC, FID, FID * FID, FID, FID, TT);
  GEMM<0, 0>(stream, nb2, P(P_PNA_LIN_W), P(P_PNA_LIN_B), nb1, NN, CC, CC, CC, CC, CC, 0, 0, 0, 0, 1);
  k_layernorm<<<NN, CC, 0, stream>>>(nb1, P(P_PNAN_G), P(P_PNAN_B), xh, nullptr, xh, CC, 2 | 4);

  // ---------- GPS ----------
  k_copy<<<GR((long)NN * CC), 256, 0, stream>>>(xh, nb2, NN * CC);        // x_in = xh
  k_gine<<<GR((long)EE * CC), 256, 0, stream>>>(xh, ea, P(P_GE_W), P(P_GE_B), src, dst, nb2);
  GEMM<0, 2>(stream, nb2, P(P_GN1_W), P(P_GN1_B), t512a, NN, 512, CC, CC, 512, 512, 0, 0, 0, 0, 1);
  GEMM<0, 0>(stream, t512a, P(P_GN2_W), P(P_GN2_B), nb3, NN, CC, 512, 512, CC, CC, 0, 0, 0, 0, 1); // x_local
  GEMM<0, 0>(stream, xh, P(P_AWQ), P(P_ABQ), nb4, NN, CC, CC, CC, CC, CC, 0, 0, 0, 0, 1);
  GEMM<0, 0>(stream, xh, P(P_AWK), P(P_ABK), nb5, NN, CC, CC, CC, CC, CC, 0, 0, 0, 0, 1);
  GEMM<0, 0>(stream, xh, P(P_AWV), P(P_ABV), nb6, NN, CC, CC, CC, CC, CC, 0, 0, 0, 0, 1);
  k_graph_attn<<<dim3(NBG, NHEADS), 32, 0, stream>>>(nb4, nb5, nb6, nb1);
  GEMM<0, 0>(stream, nb1, P(P_AWO), P(P_ABO), nb2, NN, CC, CC, CC, CC, CC, 0, 0, 0, 0, 1); // x_global
  k_concat<<<GR((long)NN * 512), 256, 0, stream>>>(nb3, nb2, t512a);
  GEMM<0, 2>(stream, t512a, P(P_CMB1_W), P(P_CMB1_B), t512b, NN, 512, 512, 512, 512, 512, 0, 0, 0, 0, 1);
  GEMM<0, 0>(stream, t512b, P(P_CMB2_W), P(P_CMB2_B), nb1, NN, CC, 512, 512, CC, CC, 0, 0, 0, 0, 1);
  k_layernorm<<<NN, CC, 0, stream>>>(nb1, P(P_GPSN_G), P(P_GPSN_B), nb2, xh, nullptr, CC, 1);
  k_layernorm<<<NN, CC, 0, stream>>>(nb2, P(P_GPSN2_G), P(P_GPSN2_B), xh, nullptr, xh, CC, 2 | 4);

  // ---------- EdgeConv (max aggregation) ----------
  GEMM<2, 3>(stream, xh, P(P_EC1_W), P(P_EC1_B), mA, EE, CC, 2 * CC,
             CC, CC, CC, 0, 0, 0, 0, 1, dst, src, nullptr, P(P_ECBN_G), P(P_ECBN_B));
  GEMM<0, 0>(stream, mA, P(P_EC2_W), P(P_EC2_B), he, EE, HD2, CC, CC, HD2, HD2, 0, 0, 0, 0, 1);
  k_fill<<<GR((long)NN * HD2), 256, 0, stream>>>(nb3, -3.0e38f, NN * HD2);
  k_edgemax<<<GR((long)EE * HD2), 256, 0, stream>>>(he, dst, nb3);
  k_layernorm<<<NN, HD2, 0, stream>>>(nb3, P(P_DN_G), P(P_DN_B), nb4, nullptr, nullptr, HD2, 2);

  // ---------- temporal attention ----------
  GEMM<0, 0>(stream, nb4, P(P_TWQ), P(P_TBQ), nb5, NN, HD2, HD2, HD2, HD2, HD2, 0, 0, 0, 0, 1);
  GEMM<0, 0>(stream, nb4, P(P_TWK), P(P_TBK), nb6, NN, HD2, HD2, HD2, HD2, HD2, 0, 0, 0, 0, 1);
  GEMM<0, 0>(stream, nb4, P(P_TWV), P(P_TBV), nb1, NN, HD2, HD2, HD2, HD2, HD2, 0, 0, 0, 0, 1);
  k_temporal_attn<<<dim3(NN, 4), 128, 0, stream>>>(nb5, nb6, nb1, nb2);
  GEMM<0, 0>(stream, nb2, P(P_TWO), P(P_TBO), nb3, NN, HD2, HD2, HD2, HD2, HD2, 0, 0, 0, 0, 1); // xa

  // ---------- value head ----------
  GEMM<0, 2>(stream, nb3, P(P_VH1_W), P(P_VH1_B), nb4, NN, 64, HD2, HD2, 64, 64, 0, 0, 0, 0, 1);
  GEMM<0, 2>(stream, nb4, P(P_VH2_W), P(P_VH2_B), nb5, NN, 32, 64, 64, 32, 32, 0, 0, 0, 0, 1);
  k_final<<<GR(NN), 256, 0, stream>>>(nb5, P(P_MEAN_W), P(P_MEAN_B), P(P_TEMP), out);
}
